// GPT_Network_82660940579118
// MI455X (gfx1250) — matrix operand order repaired and checked by reading, not compiled, not run on hardware
//
#include <hip/hip_runtime.h>
#include <hip/hip_bf16.h>
#include <cstdint>
#include <cstddef>

// ---------------- problem constants ----------------
#define L_    4
#define NH_   8
#define HID_  512
#define FF_   2048
#define V_    32000
#define S_    1024
#define E_    128
#define B_    2
#define HD_   64
#define R_    (B_ * S_)   // 2048 activation rows

typedef __attribute__((ext_vector_type(16))) __bf16 v16bf;
typedef __attribute__((ext_vector_type(8)))  float  v8f;

__device__ __forceinline__ __bf16 f2bf(float f) {
  unsigned u = __builtin_bit_cast(unsigned, f);
  unsigned r = u + 0x7FFFu + ((u >> 16) & 1u);   // round-to-nearest-even
  unsigned short h = (unsigned short)(r >> 16);
  return __builtin_bit_cast(__bf16, h);
}

__device__ __forceinline__ v8f vzero8() {
  v8f z;
#pragma unroll
  for (int i = 0; i < 8; ++i) z[i] = 0.f;
  return z;
}

// -------- CDNA5 async global->LDS copy path (ASYNCcnt-tracked) --------
#if defined(__has_builtin)
#  if __has_builtin(__builtin_amdgcn_global_load_async_to_lds_b128)
#    define USE_ASYNC_LDS 1
#  endif
#endif
#ifndef USE_ASYNC_LDS
#  define USE_ASYNC_LDS 0
#endif

#if USE_ASYNC_LDS
typedef int vint4 __attribute__((vector_size(16)));
typedef __attribute__((address_space(1))) vint4* gas_v4p;
typedef __attribute__((address_space(3))) vint4* las_v4p;

__device__ __forceinline__ void async_copy16(const void* g, void* l) {
  __builtin_amdgcn_global_load_async_to_lds_b128((gas_v4p)g, (las_v4p)l, 0, 0);
}
__device__ __forceinline__ void async_wait0() {
#if __has_builtin(__builtin_amdgcn_s_wait_asynccnt)
  __builtin_amdgcn_s_wait_asynccnt(0);
#else
  asm volatile("s_wait_asynccnt 0x0" ::: "memory");
#endif
}
#else
__device__ __forceinline__ void async_copy16(const void* g, void* l) {
  *(uint4*)l = *(const uint4*)g;
}
__device__ __forceinline__ void async_wait0() {}
#endif

// =====================================================================
// GEMM: C[M,N] = A[M,K] (bf16) * B[K,N] (bf16); fp32 accum via
// v_wmma_f32_16x16x32_bf16. Block tile 128x128, K-step 32, 8 waves,
// each wave a 32x64 C tile = 2x4 WMMA accumulators (8 WMMAs / K-step).
// Staging via global_load_async_to_lds_b128 (2 x 16B per thread per
// tile, straight-line). Optional fp32/bf16 outputs, bias, relu, scale.
// Requires M%128==0, N%128==0, K%32==0 (true for every call here).
// =====================================================================
#define BM 128
#define BN 128
#define BK 32

__global__ __launch_bounds__(256) void gemm_bf16_wmma(
    const __bf16* __restrict__ A, const __bf16* __restrict__ Bm,
    float* __restrict__ Cf, __bf16* __restrict__ Cb,
    const float* __restrict__ bias, int M, int N, int K, int relu, float scale)
{
  __shared__ __attribute__((aligned(16))) __bf16 As[BM][BK + 8];
  __shared__ __attribute__((aligned(16))) __bf16 Bs[BK][BN + 8];

  const int tid  = threadIdx.x;
  const int lane = tid & 31;
  const int wave = tid >> 5;
  const int mrow = lane & 15;
  const int hi   = lane >> 4;
  const int wm   = (wave & 3) * 32;    // 4 waves along M
  const int wn   = (wave >> 2) * 64;   // 2 waves along N
  const int bm   = blockIdx.y * BM;
  const int bn   = blockIdx.x * BN;

  // fixed per-thread staging coordinates (2 x 16B chunks per tile)
  const int arow = tid >> 2, acol = (tid & 3) * 8;    // A: 128x32, 4 chunks/row
  const int brow = tid >> 4, bcol = (tid & 15) * 8;   // B: 32x128, 16 chunks/row

  v8f acc[2][4];
#pragma unroll
  for (int i = 0; i < 2; ++i)
#pragma unroll
    for (int j = 0; j < 4; ++j) acc[i][j] = vzero8();

  for (int k0 = 0; k0 < K; k0 += BK) {
    // ---- stage A tile (straight-line, async to LDS) ----
    async_copy16(&A[(size_t)(bm + arow)      * K + (k0 + acol)], &As[arow][acol]);
    async_copy16(&A[(size_t)(bm + arow + 64) * K + (k0 + acol)], &As[arow + 64][acol]);
    // ---- stage B tile ----
    async_copy16(&Bm[(size_t)(k0 + brow)      * N + (bn + bcol)], &Bs[brow][bcol]);
    async_copy16(&Bm[(size_t)(k0 + brow + 16) * N + (bn + bcol)], &Bs[brow + 16][bcol]);
    async_wait0();
    __syncthreads();

    // prefetch next K tiles while this one computes
    if (k0 + BK < K) {
      __builtin_prefetch(&A[(size_t)(bm + (tid & 127)) * K + (k0 + BK)], 0, 1);
      __builtin_prefetch(&Bm[(size_t)(k0 + BK + (tid & 31)) * N + bn + ((tid >> 5) * 16)], 0, 1);
    }

    // A fragments (ISA 16x32 layout): VGPR j <- K pair at
    // kb = (j/4)*16 + (lane/16)*8 + (j%4)*2, row = lane&15
    v16bf afr[2], bfr[4];
#pragma unroll
    for (int t = 0; t < 2; ++t) {
#pragma unroll
      for (int j = 0; j < 8; ++j) {
        int kb = (j >> 2) * 16 + hi * 8 + (j & 3) * 2;
        afr[t][2 * j]     = As[wm + t * 16 + mrow][kb];
        afr[t][2 * j + 1] = As[wm + t * 16 + mrow][kb + 1];
      }
    }
    // B fragments (32x16): col = lane&15, VGPR j packs K pair at same kb as A
#pragma unroll
    for (int t = 0; t < 4; ++t) {
#pragma unroll
      for (int j = 0; j < 8; ++j) {
        int kb = (j >> 2) * 16 + hi * 8 + (j & 3) * 2;
        bfr[t][2 * j]     = Bs[kb][wn + t * 16 + mrow];
        bfr[t][2 * j + 1] = Bs[kb + 1][wn + t * 16 + mrow];
      }
    }

#pragma unroll
    for (int ti = 0; ti < 2; ++ti)
#pragma unroll
      for (int tj = 0; tj < 4; ++tj)
        acc[ti][tj] = __builtin_amdgcn_wmma_f32_16x16x32_bf16(
            false, afr[ti], false, bfr[tj], (short)0, acc[ti][tj], false, false);

    __syncthreads();
  }

  // epilogue: C layout — VGPR r -> row r + 8*(lane/16), col = lane&15
#pragma unroll
  for (int ti = 0; ti < 2; ++ti) {
#pragma unroll
    for (int tj = 0; tj < 4; ++tj) {
#pragma unroll
      for (int r = 0; r < 8; ++r) {
        int row = bm + wm + ti * 16 + r + 8 * hi;
        int col = bn + wn + tj * 16 + mrow;
        float v = acc[ti][tj][r] * scale;
        if (bias) v += bias[col];
        if (relu) v = fmaxf(v, 0.f);
        size_t idx = (size_t)row * N + col;
        if (Cf) Cf[idx] = v;
        if (Cb) Cb[idx] = f2bf(v);
      }
    }
  }
}

// =====================================================================
// Flash windowed causal attention, one wave per 16-query tile.
// Q,K,V,O are bf16 [B*S, HID], head h at cols h*64; Q pre-scaled 1/8.
// =====================================================================
__global__ __launch_bounds__(32) void attn_flash_wmma(
    const __bf16* __restrict__ Q, const __bf16* __restrict__ Kb,
    const __bf16* __restrict__ Vb, __bf16* __restrict__ O, int win)
{
  const int q0   = blockIdx.x * 16;
  const int h    = blockIdx.y;
  const int b    = blockIdx.z;
  const int lane = threadIdx.x;
  const int mrow = lane & 15;
  const int hi   = lane >> 4;

  __shared__ __bf16 Pl[16][32 + 8];

  v16bf qf[2];
  const size_t baseQ = ((size_t)b * S_ + q0) * HID_ + h * HD_;
#pragma unroll
  for (int t = 0; t < 2; ++t) {
#pragma unroll
    for (int j = 0; j < 8; ++j) {
      int kb = t * 32 + (j >> 2) * 16 + hi * 8 + (j & 3) * 2;
      qf[t][2 * j]     = Q[baseQ + (size_t)mrow * HID_ + kb];
      qf[t][2 * j + 1] = Q[baseQ + (size_t)mrow * HID_ + kb + 1];
    }
  }

  v8f ctx[4];
#pragma unroll
  for (int t = 0; t < 4; ++t) ctx[t] = vzero8();
  float mrun[8], lrun[8];
#pragma unroll
  for (int r = 0; r < 8; ++r) { mrun[r] = -3.0e38f; lrun[r] = 0.f; }

  int lo = q0 - win; if (lo < 0) lo = 0; lo &= ~31;

  for (int kc = lo; kc <= q0; kc += 32) {
    v8f sc[2];
    sc[0] = vzero8(); sc[1] = vzero8();
#pragma unroll
    for (int u = 0; u < 2; ++u) {         // key sub-tile
#pragma unroll
      for (int t = 0; t < 2; ++t) {       // HD half (K=32)
        v16bf kfr;
        int key = min(kc + u * 16 + mrow, S_ - 1);   // B-frag: col = key
#pragma unroll
        for (int j = 0; j < 8; ++j) {
          int hd = t * 32 + (j >> 2) * 16 + hi * 8 + (j & 3) * 2;
          kfr[2 * j]     = Kb[((size_t)b * S_ + key) * HID_ + h * HD_ + hd];
          kfr[2 * j + 1] = Kb[((size_t)b * S_ + key) * HID_ + h * HD_ + hd + 1];
        }
        sc[u] = __builtin_amdgcn_wmma_f32_16x16x32_bf16(
            false, qf[t], false, kfr, (short)0, sc[u], false, false);
      }
    }

    // online softmax (per-row stats replicated across each 16-lane half)
#pragma unroll
    for (int r = 0; r < 8; ++r) {
      int row = q0 + r + 8 * hi;
      int key0 = kc + mrow;
      int key1 = kc + 16 + mrow;
      bool v0 = (key0 <= row) && (row - key0 <= win);
      bool v1 = (key1 <= row) && (row - key1 <= win);
      float s0 = v0 ? sc[0][r] : -3.0e38f;
      float s1 = v1 ? sc[1][r] : -3.0e38f;
      float cm = fmaxf(s0, s1);
#pragma unroll
      for (int off = 1; off < 16; off <<= 1)
        cm = fmaxf(cm, __shfl_xor(cm, off));
      float mn = fmaxf(mrun[r], cm);
      float rescale = __expf(mrun[r] - mn);
      float p0 = v0 ? __expf(sc[0][r] - mn) : 0.f;
      float p1 = v1 ? __expf(sc[1][r] - mn) : 0.f;
      float rs = p0 + p1;
#pragma unroll
      for (int off = 1; off < 16; off <<= 1)
        rs += __shfl_xor(rs, off);
      mrun[r] = mn;
      lrun[r] = lrun[r] * rescale + rs;
#pragma unroll
      for (int t = 0; t < 4; ++t) ctx[t][r] *= rescale;
      Pl[r + 8 * hi][mrow]      = f2bf(p0);
      Pl[r + 8 * hi][16 + mrow] = f2bf(p1);
    }
    __syncthreads();

    v16bf pf;
#pragma unroll
    for (int j = 0; j < 8; ++j) {
      int kb = (j >> 2) * 16 + hi * 8 + (j & 3) * 2;
      pf[2 * j]     = Pl[mrow][kb];
      pf[2 * j + 1] = Pl[mrow][kb + 1];
    }
#pragma unroll
    for (int t = 0; t < 4; ++t) {
      v16bf vfr;
#pragma unroll
      for (int j = 0; j < 8; ++j) {
        int kb = (j >> 2) * 16 + hi * 8 + (j & 3) * 2;
        int k1 = min(kc + kb, S_ - 1), k2 = min(kc + kb + 1, S_ - 1);
        vfr[2 * j]     = Vb[((size_t)b * S_ + k1) * HID_ + h * HD_ + t * 16 + mrow];
        vfr[2 * j + 1] = Vb[((size_t)b * S_ + k2) * HID_ + h * HD_ + t * 16 + mrow];
      }
      ctx[t] = __builtin_amdgcn_wmma_f32_16x16x32_bf16(
          false, pf, false, vfr, (short)0, ctx[t], false, false);
    }
    __syncthreads();
  }

#pragma unroll
  for (int r = 0; r < 8; ++r) {
    float invl = 1.f / lrun[r];
    int row = q0 + r + 8 * hi;
#pragma unroll
    for (int t = 0; t < 4; ++t) {
      int col = h * HD_ + t * 16 + mrow;
      O[((size_t)b * S_ + row) * HID_ + col] = f2bf(ctx[t][r] * invl);
    }
  }
}

// =====================================================================
// out = res + LayerNorm(y)*scale + bias ; optional bf16 copy of out
// =====================================================================
__global__ __launch_bounds__(256) void resid_ln(
    const float* __restrict__ y, const float* __restrict__ res,
    const float* __restrict__ bias, const float* __restrict__ scale,
    float* __restrict__ out, __bf16* __restrict__ outb, int res_broadcast)
{
  __shared__ float red[8];
  const int row = blockIdx.x;
  const int s   = row % S_;
  const int tid = threadIdx.x;
  const int lane = tid & 31, wave = tid >> 5;
  const float* yr = y + (size_t)row * HID_;
  const float* rr = res + (size_t)(res_broadcast ? s : row) * HID_;

  float sum = 0.f;
  for (int i = tid; i < HID_; i += 256) sum += yr[i];
#pragma unroll
  for (int off = 16; off > 0; off >>= 1) sum += __shfl_xor(sum, off);
  if (lane == 0) red[wave] = sum;
  __syncthreads();
  float tot = 0.f;
#pragma unroll
  for (int i = 0; i < 8; ++i) tot += red[i];
  const float mean = tot * (1.f / HID_);
  __syncthreads();

  float vs = 0.f;
  for (int i = tid; i < HID_; i += 256) { float d = yr[i] - mean; vs += d * d; }
#pragma unroll
  for (int off = 16; off > 0; off >>= 1) vs += __shfl_xor(vs, off);
  if (lane == 0) red[wave] = vs;
  __syncthreads();
  float vtot = 0.f;
#pragma unroll
  for (int i = 0; i < 8; ++i) vtot += red[i];
  const float inv = rsqrtf(vtot * (1.f / HID_) + 1e-6f);

  for (int i = tid; i < HID_; i += 256) {
    float v = rr[i] + (yr[i] - mean) * inv * scale[i] + bias[i];
    out[(size_t)row * HID_ + i] = v;
    if (outb) outb[(size_t)row * HID_ + i] = f2bf(v);
  }
}

// embedding gather straight to bf16
__global__ __launch_bounds__(128) void embed_gather(
    const int* __restrict__ x, const float* __restrict__ Wemb,
    __bf16* __restrict__ out)
{
  int row = blockIdx.x;
  int e   = threadIdx.x;
  out[(size_t)row * E_ + e] = f2bf(Wemb[(size_t)x[row] * E_ + e]);
}

// fp32 -> bf16 tensor conversion (weights, once per launch)
__global__ __launch_bounds__(256) void cvt_f32_bf16(
    const float* __restrict__ in, __bf16* __restrict__ out, long n)
{
  long i = (long)blockIdx.x * 256 + threadIdx.x;
  long stride = (long)gridDim.x * 256;
  for (; i < n; i += stride) out[i] = f2bf(in[i]);
}

// =====================================================================
// host orchestration
// =====================================================================
static inline void launch_gemm(const __bf16* A, const __bf16* Bm, float* Cf,
                               __bf16* Cb, const float* bias, int M, int N,
                               int K, int relu, float scale, hipStream_t st) {
  dim3 grid(N / BN, M / BM);
  gemm_bf16_wmma<<<grid, 256, 0, st>>>(A, Bm, Cf, Cb, bias, M, N, K, relu, scale);
}

extern "C" void kernel_launch(void* const* d_in, const int* in_sizes, int n_in,
                              void* d_out, int out_size, void* d_ws, size_t ws_size,
                              hipStream_t stream) {
  (void)in_sizes; (void)n_in; (void)out_size; (void)ws_size;
  const int*   x_input    = (const int*)  d_in[0];
  const float* W_emb_dec  = (const float*)d_in[1];
  const float* W_dec_lin  = (const float*)d_in[2];
  const float* p_decoder  = (const float*)d_in[3];
  const float* p_d_q      = (const float*)d_in[4];
  const float* p_d_k      = (const float*)d_in[5];
  const float* p_d_v      = (const float*)d_in[6];
  const float* p_d_c      = (const float*)d_in[7];
  const float* p_d_ff1    = (const float*)d_in[8];
  const float* p_d_ff2    = (const float*)d_in[9];
  const float* b_d_ff1    = (const float*)d_in[10];
  const float* b_d_ff2    = (const float*)d_in[11];
  const float* b_d_bias_i = (const float*)d_in[12];
  const float* b_d_scale_i= (const float*)d_in[13];
  const float* b_d_bias_1 = (const float*)d_in[14];
  const float* b_d_scale_1= (const float*)d_in[15];
  const float* b_d_bias_2 = (const float*)d_in[16];
  const float* b_d_scale_2= (const float*)d_in[17];
  const float* d_o_bias   = (const float*)d_in[18];
  const float* d_o_scale  = (const float*)d_in[19];
  const float* x_emb_pos  = (const float*)d_in[20];
  float* logits = (float*)d_out;

  char* base = (char*)d_ws;
  size_t off = 0;
  auto allocf = [&](size_t n) {
    float* p = (float*)(base + off); off += ((n * 4 + 255) & ~(size_t)255); return p; };
  auto allocb = [&](size_t n) {
    __bf16* p = (__bf16*)(base + off); off += ((n * 2 + 255) & ~(size_t)255); return p; };

  // bf16 weights
  __bf16* wlin_b = allocb((size_t)E_ * HID_);
  __bf16* wq_b   = allocb((size_t)L_ * HID_ * HID_);
  __bf16* wk_b   = allocb((size_t)L_ * HID_ * HID_);
  __bf16* wv_b   = allocb((size_t)L_ * HID_ * HID_);
  __bf16* wc_b   = allocb((size_t)L_ * HID_ * HID_);
  __bf16* wf1_b  = allocb((size_t)L_ * HID_ * FF_);
  __bf16* wf2_b  = allocb((size_t)L_ * FF_ * HID_);
  __bf16* wdec_b = allocb((size_t)HID_ * V_);
  // activations
  __bf16* emb_b      = allocb((size_t)R_ * E_);
  float*  dec_in_f   = allocf((size_t)R_ * HID_);
  float*  layer_in_f = allocf((size_t)R_ * HID_);
  __bf16* layer_in_b = allocb((size_t)R_ * HID_);
  __bf16* q_b        = allocb((size_t)R_ * HID_);
  __bf16* k_b        = allocb((size_t)R_ * HID_);
  __bf16* v_b        = allocb((size_t)R_ * HID_);
  __bf16* ctx_b      = allocb((size_t)R_ * HID_);
  float*  attn_out_f = allocf((size_t)R_ * HID_);
  float*  x1_f       = allocf((size_t)R_ * HID_);
  __bf16* x1_b       = allocb((size_t)R_ * HID_);
  __bf16* h_b        = allocb((size_t)R_ * FF_);
  float*  ffw_f      = allocf((size_t)R_ * HID_);
  float*  x_f        = allocf((size_t)R_ * HID_);
  float*  dec_out_f  = allocf((size_t)R_ * HID_);
  __bf16* dec_out_b  = allocb((size_t)R_ * HID_);

  auto cvt = [&](const float* in, __bf16* out, size_t n) {
    size_t g = (n + 255) / 256; if (g > 4096) g = 4096;
    cvt_f32_bf16<<<(int)g, 256, 0, stream>>>(in, out, (long)n);
  };

  // one-time weight downcast (bf16 operands for every WMMA GEMM)
  cvt(W_dec_lin, wlin_b, (size_t)E_ * HID_);
  cvt(p_d_q,  wq_b,  (size_t)L_ * HID_ * HID_);
  cvt(p_d_k,  wk_b,  (size_t)L_ * HID_ * HID_);
  cvt(p_d_v,  wv_b,  (size_t)L_ * HID_ * HID_);
  cvt(p_d_c,  wc_b,  (size_t)L_ * HID_ * HID_);
  cvt(p_d_ff1, wf1_b, (size_t)L_ * HID_ * FF_);
  cvt(p_d_ff2, wf2_b, (size_t)L_ * FF_ * HID_);
  cvt(p_decoder, wdec_b, (size_t)HID_ * V_);

  static const int WIN[L_] = {256, 341, 512, 1024};

  embed_gather<<<R_, 128, 0, stream>>>(x_input, W_emb_dec, emb_b);
  launch_gemm(emb_b, wlin_b, dec_in_f, nullptr, nullptr, R_, HID_, E_, 0, 1.f, stream);

  const float* cur = dec_in_f;
  for (int m = 0; m < L_; ++m) {
    resid_ln<<<R_, 256, 0, stream>>>(cur, x_emb_pos + (size_t)m * S_ * HID_,
                                     b_d_bias_i + m * HID_, b_d_scale_i + m * HID_,
                                     layer_in_f, layer_in_b, 1);
    // Q scaled by 1/sqrt(HD) in the epilogue; K,V plain
    launch_gemm(layer_in_b, wq_b + (size_t)m * HID_ * HID_, nullptr, q_b,
                nullptr, R_, HID_, HID_, 0, 0.125f, stream);
    launch_gemm(layer_in_b, wk_b + (size_t)m * HID_ * HID_, nullptr, k_b,
                nullptr, R_, HID_, HID_, 0, 1.f, stream);
    launch_gemm(layer_in_b, wv_b + (size_t)m * HID_ * HID_, nullptr, v_b,
                nullptr, R_, HID_, HID_, 0, 1.f, stream);
    dim3 agrid(S_ / 16, NH_, B_);
    attn_flash_wmma<<<agrid, 32, 0, stream>>>(q_b, k_b, v_b, ctx_b, WIN[m]);
    launch_gemm(ctx_b, wc_b + (size_t)m * HID_ * HID_, attn_out_f, nullptr,
                nullptr, R_, HID_, HID_, 0, 1.f, stream);
    resid_ln<<<R_, 256, 0, stream>>>(attn_out_f, layer_in_f,
                                     b_d_bias_1 + m * HID_, b_d_scale_1 + m * HID_,
                                     x1_f, x1_b, 0);
    launch_gemm(x1_b, wf1_b + (size_t)m * HID_ * FF_, nullptr, h_b,
                b_d_ff1 + m * FF_, R_, FF_, HID_, 1, 1.f, stream);
    launch_gemm(h_b, wf2_b + (size_t)m * FF_ * HID_, ffw_f, nullptr,
                b_d_ff2 + m * HID_, R_, HID_, FF_, 0, 1.f, stream);
    resid_ln<<<R_, 256, 0, stream>>>(ffw_f, x1_f,
                                     b_d_bias_2 + m * HID_, b_d_scale_2 + m * HID_,
                                     x_f, nullptr, 0);
    cur = x_f;
  }

  resid_ln<<<R_, 256, 0, stream>>>(cur, dec_in_f, d_o_bias, d_o_scale,
                                   dec_out_f, dec_out_b, 0);
  launch_gemm(dec_out_b, wdec_b, logits, nullptr, nullptr, R_, V_, HID_, 0, 1.f, stream);
}
